// SlotAttentionVideo_85272280694999
// MI455X (gfx1250) — matrix operand order repaired and checked by reading, not compiled, not run on hardware
//
#include <hip/hip_runtime.h>
#include <hip/hip_bf16.h>

typedef __attribute__((ext_vector_type(16))) __bf16 v16bf;
typedef __attribute__((ext_vector_type(8)))  __bf16 v8bf;
typedef __attribute__((ext_vector_type(8)))  float  v8f;

namespace {
constexpr int Bn = 32, Tn = 16, Nn = 256, Cn = 192, Sn = 8, Dn = 192;
constexpr float SCALEk = 0.07216878364870323f; // 192^-0.5
constexpr float EPSk = 1e-8f;
}

__device__ __forceinline__ v8f wmma_bf16(v16bf a, v16bf b, v8f c) {
  // D = A(16x32 bf16) * B(32x16 bf16) + C(16x16 f32)
  return __builtin_amdgcn_wmma_f32_16x16x32_bf16(false, a, false, b, (short)0, c, false, false);
}

// Cooperative LayerNorm of a 192-float row by one wave32.
__device__ __forceinline__ void ln_row192(const float* __restrict__ in, float* __restrict__ out,
                                          const float* __restrict__ g, const float* __restrict__ bb,
                                          int lane) {
  float v[6], s = 0.f, s2 = 0.f;
#pragma unroll
  for (int j = 0; j < 6; ++j) { float t = in[lane + 32*j]; v[j] = t; s += t; s2 += t*t; }
#pragma unroll
  for (int off = 16; off; off >>= 1) { s += __shfl_xor(s, off, 32); s2 += __shfl_xor(s2, off, 32); }
  const float mu  = s * (1.f/192.f);
  const float inv = rsqrtf(s2 * (1.f/192.f) - mu*mu + 1e-5f);
#pragma unroll
  for (int j = 0; j < 6; ++j) { const int c = lane + 32*j; out[c] = (v[j]-mu)*inv*g[c] + bb[c]; }
}

// Dense layer via WMMA: out[8][Nout](f32 LDS) = act[8][K](f32 LDS) @ Wbf[K][Nout](bf16 global)
// M=16 tile: slots in rows 0..7 (lanes' M), rows 8..15 padded with zeros.
// mode: 0 = store (+bias), 1 = relu(store+bias), 2 = accumulate (+bias)
__device__ __forceinline__ void dense8_wmma(const float* __restrict__ act, int K,
                                            const __bf16* __restrict__ Wbf, int Nout,
                                            const float* __restrict__ bias,
                                            float* __restrict__ out,
                                            int w, int lane, int mode) {
  const int ntiles = Nout >> 4;
  const int ksteps = K >> 5;
  const int sl = lane & 15, si = sl & 7;
  const int halfk = (lane >> 4) * 8;
  for (int nt = w; nt < ntiles; nt += 16) {   // wave-uniform: EXEC stays all-ones
    v8f c = {};
    for (int ks = 0; ks < ksteps; ++ks) {
      v16bf a;
#pragma unroll
      for (int e = 0; e < 16; ++e) {
        const int kidx = ks*32 + ((e >> 3) << 4) + halfk + (e & 7);
        const float av = act[si*K + kidx];
        a[e] = (sl < 8) ? (__bf16)av : (__bf16)0.f;
      }
      v16bf bf;
#pragma unroll
      for (int e = 0; e < 16; ++e)
        bf[e] = Wbf[(size_t)(ks*32 + ((e >> 3) << 4) + halfk + (e & 7))*Nout + nt*16 + sl];
      c = wmma_bf16(a, bf, c);
    }
    if (lane < 16) {                          // D: lanes 0..15 hold rows m=0..7
#pragma unroll
      for (int r = 0; r < 8; ++r) {
        const int col = nt*16 + lane;
        float v = c[r];
        if (bias) v += bias[col];
        if (mode == 1) v = fmaxf(v, 0.f);
        if (mode == 2) out[r*Nout + col] += v;
        else           out[r*Nout + col]  = v;
      }
    }
  }
}

// ---------------- kernel 0: convert / transpose all weights to bf16 ----------------
__global__ __launch_bounds__(256) void cvt_all_kernel(
    const float* __restrict__ Wk, const float* __restrict__ Wv, const float* __restrict__ Wq,
    const float* __restrict__ pw1, const float* __restrict__ pw2,
    const float* __restrict__ wih, const float* __restrict__ whh,
    const float* __restrict__ w1, const float* __restrict__ w2,
    __bf16* __restrict__ Wkb, __bf16* __restrict__ Wvb, __bf16* __restrict__ Wqb,
    __bf16* __restrict__ pw1b, __bf16* __restrict__ pw2b,
    __bf16* __restrict__ wihTb, __bf16* __restrict__ whhTb,
    __bf16* __restrict__ w1b, __bf16* __restrict__ w2b)
{
  const int i = blockIdx.x * 256 + threadIdx.x;
  if (i < Cn*Dn) {
    Wkb[i]  = (__bf16)Wk[i];  Wvb[i]  = (__bf16)Wv[i];  Wqb[i] = (__bf16)Wq[i];
    pw1b[i] = (__bf16)pw1[i]; pw2b[i] = (__bf16)pw2[i];
  }
  if (i < 3*Dn*Dn) {               // gru weights: [3D][D] -> transposed [D][3D]
    const int j = i / Dn, d = i % Dn;
    wihTb[d*3*Dn + j] = (__bf16)wih[i];
    whhTb[d*3*Dn + j] = (__bf16)whh[i];
  }
  if (i < Dn*4*Dn) {               // mlp weights already [K][N] row-major
    w1b[i] = (__bf16)w1[i];
    w2b[i] = (__bf16)w2[i];
  }
}

// ---------------- kernel 1: LN + k/v projection (WMMA bf16) ----------------
// 128 rows per block; 8 waves, one 16-row M-tile per wave; K=192 (6 steps), N=192 (12 tiles).
__global__ __launch_bounds__(256) void ln_kv_kernel(
    const float* __restrict__ xin, const float* __restrict__ g, const float* __restrict__ bta,
    const __bf16* __restrict__ Wk_bf, const __bf16* __restrict__ Wv_bf,
    __bf16* __restrict__ k_out, __bf16* __restrict__ v_out)
{
  extern __shared__ char smem[];
  __bf16* xs  = (__bf16*)smem;                  // [128][192] LN'd activations, bf16
  __bf16* wsh = (__bf16*)(smem + 128*Cn*2);     // [192][192] staged weights, bf16
  const int tid = threadIdx.x, w = tid >> 5, lane = tid & 31;
  const long row0 = (long)blockIdx.x * 128;

  for (int i = 0; i < 16; ++i) {
    const int r = w * 16 + i;
    const float* xp = xin + (row0 + r) * Cn;
    float vals[6], s = 0.f, s2 = 0.f;
#pragma unroll
    for (int j = 0; j < 6; ++j) { float t = xp[lane + 32*j]; vals[j] = t; s += t; s2 += t*t; }
#pragma unroll
    for (int off = 16; off; off >>= 1) { s += __shfl_xor(s, off, 32); s2 += __shfl_xor(s2, off, 32); }
    const float mu  = s * (1.f/Cn);
    const float inv = rsqrtf(s2 * (1.f/Cn) - mu*mu + 1e-5f);
#pragma unroll
    for (int j = 0; j < 6; ++j) {
      const int c = lane + 32*j;
      xs[r*Cn + c] = (__bf16)((vals[j]-mu)*inv*g[c] + bta[c]);
    }
  }
  __syncthreads();

  v16bf afr[6];
  const int m = w*16 + (lane & 15);
  const int halfk = (lane >> 4) * 8;
#pragma unroll
  for (int ks = 0; ks < 6; ++ks) {
    const __bf16* base = &xs[m*Cn + ks*32 + halfk];
    v8bf lo = *(const v8bf*)base;
    v8bf hi = *(const v8bf*)(base + 16);
#pragma unroll
    for (int e = 0; e < 8; ++e) { afr[ks][e] = lo[e]; afr[ks][8+e] = hi[e]; }
  }

  for (int pass = 0; pass < 2; ++pass) {
    const __bf16* Wg = pass ? Wv_bf : Wk_bf;
    __bf16* outp    = pass ? v_out  : k_out;
    const float scale = pass ? 1.f : SCALEk;
    __syncthreads();
    for (int i = tid; i < (Cn*Dn)/8; i += 256)
      ((v8bf*)wsh)[i] = ((const v8bf*)Wg)[i];
    __syncthreads();
    for (int nt = 0; nt < 12; ++nt) {
      v8f c = {};
#pragma unroll
      for (int ks = 0; ks < 6; ++ks) {
        v16bf bfr;
#pragma unroll
        for (int e = 0; e < 16; ++e)
          bfr[e] = wsh[(ks*32 + ((e >> 3) << 4) + halfk + (e & 7))*Dn + nt*16 + (lane & 15)];
        c = wmma_bf16(afr[ks], bfr, c);
      }
      const int  n  = nt*16 + (lane & 15);
      const long rb = row0 + w*16 + ((lane >> 4) * 8);
#pragma unroll
      for (int r = 0; r < 8; ++r)
        outp[(rb + r)*Dn + n] = (__bf16)(c[r] * scale);
    }
  }
}

// ---------------- kernel 2: slot attention scan (one block per batch, 16 waves) ----------------
__global__ __launch_bounds__(512) void slot_scan_kernel(
    const float* __restrict__ noise, const float* __restrict__ mu0, const float* __restrict__ lsig,
    const float* __restrict__ lns_g, const float* __restrict__ lns_b,
    const float* __restrict__ bih, const float* __restrict__ bhh,
    const float* __restrict__ lnf_g, const float* __restrict__ lnf_b,
    const float* __restrict__ b1, const float* __restrict__ b2,
    const float* __restrict__ pb1, const float* __restrict__ pb2,
    const __bf16* __restrict__ Wqb,
    const __bf16* __restrict__ wihTb, const __bf16* __restrict__ whhTb,
    const __bf16* __restrict__ w1b, const __bf16* __restrict__ w2b,
    const __bf16* __restrict__ pw1b, const __bf16* __restrict__ pw2b,
    const __bf16* __restrict__ k_bf, const __bf16* __restrict__ v_bf,
    float* __restrict__ out_slots, float* __restrict__ out_dots)
{
  extern __shared__ char smem[];
  float*  slots = (float*)(smem);            // [8][192]
  float*  sprev = (float*)(smem + 6144);     // [8][192]
  float*  lnb   = (float*)(smem + 12288);    // [8][192]
  float*  upd   = (float*)(smem + 18432);    // [8][192]
  float*  qf    = (float*)(smem + 24576);    // [8][192]
  float*  dotsS = (float*)(smem + 30720);    // [256][8]
  float*  attnS = (float*)(smem + 38912);    // [256][8]
  float*  colp  = (float*)(smem + 47104);    // [32][8]
  float*  gx    = (float*)(smem + 48128);    // [8][576]
  float*  gh    = (float*)(smem + 66560);    // [8][576]
  float*  h1    = (float*)(smem + 84992);    // [8][768]  (total 109568 B)

  const int tid = threadIdx.x, w = tid >> 5, lane = tid & 31;
  const int b = blockIdx.x;

  for (int i = tid; i < Sn*Dn; i += 512) {
    const int d = i % Dn;
    slots[i] = mu0[d] + __expf(lsig[d]) * noise[b*Sn*Dn + i];
  }
  __syncthreads();

  for (int t = 0; t < Tn; ++t) {
    const __bf16* kt = k_bf + (long)(b*Tn + t) * Nn * Dn;
    const __bf16* vt = v_bf + (long)(b*Tn + t) * Nn * Dn;

    for (int it = 0; it < 3; ++it) {
      for (int i = tid; i < Sn*Dn; i += 512) sprev[i] = slots[i];
      __syncthreads();
      if (w < 8) ln_row192(slots + w*Dn, lnb + w*Dn, lns_g, lns_b, lane);
      __syncthreads();
      // q = LN(slots) @ Wq   (WMMA)
      dense8_wmma(lnb, Dn, Wqb, Dn, nullptr, qf, w, lane, 0);
      __syncthreads();
      // dots[n][s] = sum_d k[n,d]*q[s,d] : 16 token M-tiles, wave w owns tile w
      {
        const int mt = w;
        const int mrow  = mt*16 + (lane & 15);
        const int halfk = (lane >> 4) * 8;
        v8f c = {};
#pragma unroll
        for (int ks = 0; ks < 6; ++ks) {
          const __bf16* base = kt + mrow*Dn + ks*32 + halfk;
          v8bf lo = *(const v8bf*)base;
          v8bf hi = *(const v8bf*)(base + 16);
          v16bf a;
#pragma unroll
          for (int e = 0; e < 8; ++e) { a[e] = lo[e]; a[8+e] = hi[e]; }
          v16bf bf;
#pragma unroll
          for (int e = 0; e < 16; ++e) {
            const int kk = ks*32 + ((e >> 3) << 4) + halfk + (e & 7);
            bf[e] = ((lane & 15) < 8) ? (__bf16)qf[(lane & 15)*Dn + kk] : (__bf16)0.f;
          }
          c = wmma_bf16(a, bf, c);
        }
        const int sl = lane & 15;
        if (sl < 8) {
          const int rb = mt*16 + (lane >> 4) * 8;
#pragma unroll
          for (int r = 0; r < 8; ++r) dotsS[(rb + r)*Sn + sl] = c[r];
        }
      }
      __syncthreads();
      // softmax over slots per token (+ emit out_dots on last iteration)
      if (tid < Nn) {
        float vv[8], mx = -3.0e38f;
#pragma unroll
        for (int s = 0; s < 8; ++s) { vv[s] = dotsS[tid*Sn + s]; mx = fmaxf(mx, vv[s]); }
        float sum = 0.f;
#pragma unroll
        for (int s = 0; s < 8; ++s) { vv[s] = __expf(vv[s] - mx); sum += vv[s]; }
        const float inv = 1.f / sum;
#pragma unroll
        for (int s = 0; s < 8; ++s) {
          const float sm = vv[s] * inv;
          if (it == 2) out_dots[((long)(b*Tn + t)*Nn + tid)*Sn + s] = sm;
          attnS[tid*Sn + s] = sm + EPSk;
        }
      }
      __syncthreads();
      // column sums over tokens, then normalize
      if (tid < 256) {
        const int gi = tid >> 3, s = tid & 7;
        float ss = 0.f;
#pragma unroll
        for (int j = 0; j < 8; ++j) ss += attnS[(gi*8 + j)*Sn + s];
        colp[gi*8 + s] = ss;
      }
      __syncthreads();
      if (tid < 8) {
        float ss = 0.f;
        for (int gi = 0; gi < 32; ++gi) ss += colp[gi*8 + tid];
        colp[tid] = 1.f / ss;
      }
      __syncthreads();
      for (int i = tid; i < Nn*Sn; i += 512) attnS[i] *= colp[i & 7];
      __syncthreads();
      // updates[s][d] = sum_n attn[n,s]*v[n,d] : M=16(S pad), K=256, N=192 (12 tiles)
      if (w < 12) {
        const int nt = w;
        const int sl = lane & 15;
        const int si = sl & 7;
        const int halfk = (lane >> 4) * 8;
        v8f c = {};
        for (int ks = 0; ks < 8; ++ks) {
          v16bf a;
#pragma unroll
          for (int e = 0; e < 16; ++e) {
            const int tok = ks*32 + ((e >> 3) << 4) + halfk + (e & 7);
            const float av = attnS[tok*Sn + si];
            a[e] = (sl < 8) ? (__bf16)av : (__bf16)0.f;
          }
          v16bf bf;
#pragma unroll
          for (int e = 0; e < 16; ++e)
            bf[e] = vt[(ks*32 + ((e >> 3) << 4) + halfk + (e & 7))*Dn + nt*16 + sl];
          c = wmma_bf16(a, bf, c);
        }
        if (lane < 16) {
#pragma unroll
          for (int r = 0; r < 8; ++r) upd[r*Dn + nt*16 + lane] = c[r];
        }
      }
      __syncthreads();
      // GRU gates (WMMA): gx = upd@wih^T + bih ; gh = sprev@whh^T + bhh
      dense8_wmma(upd,   Dn, wihTb, 3*Dn, bih, gx, w, lane, 0);
      dense8_wmma(sprev, Dn, whhTb, 3*Dn, bhh, gh, w, lane, 0);
      __syncthreads();
      for (int i = tid; i < Sn*Dn; i += 512) {
        const int s = i / Dn, d = i % Dn;
        const float xr = gx[s*3*Dn + d], xz = gx[s*3*Dn + Dn + d], xn = gx[s*3*Dn + 2*Dn + d];
        const float hr = gh[s*3*Dn + d], hz = gh[s*3*Dn + Dn + d], hn = gh[s*3*Dn + 2*Dn + d];
        const float r_ = 1.f / (1.f + __expf(-(xr + hr)));
        const float z_ = 1.f / (1.f + __expf(-(xz + hz)));
        const float n_ = tanhf(xn + r_*hn);
        slots[i] = (1.f - z_)*n_ + z_*sprev[i];
      }
      __syncthreads();
      if (it < 2) {  // residual MLP (WMMA)
        if (w < 8) ln_row192(slots + w*Dn, lnb + w*Dn, lnf_g, lnf_b, lane);
        __syncthreads();
        dense8_wmma(lnb, Dn, w1b, 4*Dn, b1, h1, w, lane, 1);       // relu
        __syncthreads();
        dense8_wmma(h1, 4*Dn, w2b, Dn, b2, slots, w, lane, 2);     // accumulate
        __syncthreads();
      }
    } // it

    for (int i = tid; i < Sn*Dn; i += 512)
      out_slots[(long)(b*Tn + t)*Sn*Dn + i] = slots[i];
    __syncthreads();
    // prior MLP (WMMA) -> carry for next timestep
    dense8_wmma(slots, Dn, pw1b, Dn, pb1, lnb, w, lane, 1);        // relu
    __syncthreads();
    dense8_wmma(lnb, Dn, pw2b, Dn, pb2, upd, w, lane, 0);
    __syncthreads();
    for (int i = tid; i < Sn*Dn; i += 512) slots[i] = upd[i];
    __syncthreads();
  } // t
}

extern "C" void kernel_launch(void* const* d_in, const int* in_sizes, int n_in,
                              void* d_out, int out_size, void* d_ws, size_t ws_size,
                              hipStream_t stream) {
  (void)in_sizes; (void)n_in; (void)out_size; (void)ws_size;
  const float* inputs          = (const float*)d_in[0];
  const float* noise           = (const float*)d_in[1];
  const float* slots_mu        = (const float*)d_in[2];
  const float* slots_log_sigma = (const float*)d_in[3];
  const float* ln_in_g         = (const float*)d_in[4];
  const float* ln_in_b         = (const float*)d_in[5];
  const float* Wk              = (const float*)d_in[6];
  const float* Wv              = (const float*)d_in[7];
  const float* Wq              = (const float*)d_in[8];
  const float* ln_slots_g      = (const float*)d_in[9];
  const float* ln_slots_b      = (const float*)d_in[10];
  const float* gru_wih         = (const float*)d_in[11];
  const float* gru_whh         = (const float*)d_in[12];
  const float* gru_bih         = (const float*)d_in[13];
  const float* gru_bhh         = (const float*)d_in[14];
  const float* ln_ff_g         = (const float*)d_in[15];
  const float* ln_ff_b         = (const float*)d_in[16];
  const float* mlp_w1          = (const float*)d_in[17];
  const float* mlp_b1          = (const float*)d_in[18];
  const float* mlp_w2          = (const float*)d_in[19];
  const float* mlp_b2          = (const float*)d_in[20];
  const float* pr_w1           = (const float*)d_in[21];
  const float* pr_b1           = (const float*)d_in[22];
  const float* pr_w2           = (const float*)d_in[23];
  const float* pr_b2           = (const float*)d_in[24];

  char* wsb = (char*)d_ws;
  const size_t KV = (size_t)Bn * Tn * Nn * Dn;        // 25,165,824 elements
  __bf16* k_bf  = (__bf16*)wsb;
  __bf16* v_bf  = (__bf16*)(wsb + KV*2);
  size_t off = 2*KV*2;
  __bf16* Wkb   = (__bf16*)(wsb + off); off += (size_t)Cn*Dn*2;      // 73728 each
  __bf16* Wvb   = (__bf16*)(wsb + off); off += (size_t)Cn*Dn*2;
  __bf16* Wqb   = (__bf16*)(wsb + off); off += (size_t)Dn*Dn*2;
  __bf16* pw1b  = (__bf16*)(wsb + off); off += (size_t)Dn*Dn*2;
  __bf16* pw2b  = (__bf16*)(wsb + off); off += (size_t)Dn*Dn*2;
  __bf16* wihTb = (__bf16*)(wsb + off); off += (size_t)Dn*3*Dn*2;    // 221184
  __bf16* whhTb = (__bf16*)(wsb + off); off += (size_t)Dn*3*Dn*2;
  __bf16* w1b   = (__bf16*)(wsb + off); off += (size_t)Dn*4*Dn*2;    // 294912
  __bf16* w2b   = (__bf16*)(wsb + off); off += (size_t)Dn*4*Dn*2;

  float* out_slots = (float*)d_out;                           // [B,T,S,D]
  float* out_dots  = out_slots + (size_t)Bn*Tn*Sn*Dn;         // [B,T,N,S]

  cvt_all_kernel<<<(Dn*4*Dn + 255) / 256, 256, 0, stream>>>(
      Wk, Wv, Wq, pr_w1, pr_w2, gru_wih, gru_whh, mlp_w1, mlp_w2,
      Wkb, Wvb, Wqb, pw1b, pw2b, wihTb, whhTb, w1b, w2b);
  ln_kv_kernel<<<(Bn*Tn*Nn) / 128, 256, 128*Cn*2 + Cn*Dn*2, stream>>>(
      inputs, ln_in_g, ln_in_b, Wkb, Wvb, k_bf, v_bf);
  slot_scan_kernel<<<Bn, 512, 109568, stream>>>(
      noise, slots_mu, slots_log_sigma,
      ln_slots_g, ln_slots_b, gru_bih, gru_bhh, ln_ff_g, ln_ff_b,
      mlp_b1, mlp_b2, pr_b1, pr_b2,
      Wqb, wihTb, whhTb, w1b, w2b, pw1b, pw2b,
      k_bf, v_bf, out_slots, out_dots);
}
